// CausalSelfAttention_27264452395689
// MI455X (gfx1250) — matrix operand order repaired and checked by reading, not compiled, not run on hardware
//
#include <hip/hip_runtime.h>
#include <hip/hip_bf16.h>
#include <stdint.h>

// ---------------------------------------------------------------------------
// Causal self-attention, bf16 WMMA + TDM pipeline for gfx1250 (MI455X)
//   B=4, T=2048, C=1024, H=16, HD=64
// ---------------------------------------------------------------------------

#define B_  4
#define T_  2048
#define C_  1024
#define H_  16
#define HD_ 64

typedef __bf16 bf16_t;
typedef bf16_t v16bf __attribute__((ext_vector_type(16)));
typedef float  v8f   __attribute__((ext_vector_type(8)));
typedef unsigned int v4u __attribute__((ext_vector_type(4)));
typedef int v4i __attribute__((ext_vector_type(4)));
typedef int v8i __attribute__((ext_vector_type(8)));
typedef __attribute__((address_space(3))) unsigned short lds_ushort;

union FragBF {
  v16bf v;
  uint4 q[2];
};

__device__ __forceinline__ unsigned short f32_to_bf16(float f) {
  union { float f; unsigned u; } x; x.f = f;
  unsigned r = x.u + 0x7fffu + ((x.u >> 16) & 1u);
  return (unsigned short)(r >> 16);
}

// TDM: DMA a 64-row x 32-col bf16 tile of a [1024,1024] row-major matrix
// into LDS (row-major [64][32], 4 KB). Issued by one wave; TENSORcnt tracked.
__device__ __forceinline__ void tdm_load_w_tile(const unsigned short* gsrc,
                                                unsigned lds_off) {
  uint64_t ga = (uint64_t)(uintptr_t)gsrc;
  v4u g0 = { 1u,                                   // count=1, user descriptor
             lds_off,                              // lds_addr (bytes)
             (unsigned)(ga & 0xffffffffu),         // global_addr[31:0]
             (unsigned)((ga >> 32) & 0x01ffffffu)  // global_addr[56:32]
               | 0x80000000u };                    // type=2 ("image")
  v8i g1 = { (int)0x00010000u,                     // data_size=1 -> 2 bytes
             (int)(1024u << 16),                   // tensor_dim0[15:0]=1024
             (int)(1024u << 16),                   // dim0 hi=0 | tensor_dim1[15:0]=1024
             (int)(32u << 16),                     // dim1 hi=0 | tile_dim0=32
             64,                                   // tile_dim1=64, tile_dim2=0
             1024,                                 // tensor_dim0_stride[31:0]
             0, 0 };                               // stride hi, dim1_stride
  v4i gz = { 0, 0, 0, 0 };
#if defined(__clang_major__) && (__clang_major__ >= 23)
  v8i gz8 = { 0, 0, 0, 0, 0, 0, 0, 0 };
  __builtin_amdgcn_tensor_load_to_lds(g0, g1, gz, gz, gz8, 0);
#else
  __builtin_amdgcn_tensor_load_to_lds(g0, g1, gz, gz, 0);
#endif
}

// ---------------------------------------------------------------------------
// Kernel 0: f32 -> bf16 conversion (grid-stride)
// ---------------------------------------------------------------------------
__global__ void cvt_f32_bf16(const float* __restrict__ src,
                             unsigned short* __restrict__ dst, int n) {
  int i = blockIdx.x * blockDim.x + threadIdx.x;
  int stride = gridDim.x * blockDim.x;
  for (; i < n; i += stride) dst[i] = f32_to_bf16(src[i]);
}

// ---------------------------------------------------------------------------
// Kernel 1/3: NT GEMM  Y[m,n] = sum_k A[m,k] * W[n,k] + bias[n]
//   A: bf16 [8192,1024]  W: bf16 [1024,1024], both K-contiguous.
// W tile (64x32) staged in LDS by TDM, double-buffered; A fragments
// double-buffered in registers; k-loop unrolled by 2 so all buffer
// selections are compile-time constants (no scratch, no cndmask).
// Wave tile 32x64 (8 WMMA / k-step), 8 waves => block tile 256x64.
// grid = (32, 16).
// mode 0: bf16 out [B,H,T,HD]; mode 1: bf16 out [B,H,HD,T]; mode 2: f32 [M,C]
// ---------------------------------------------------------------------------
__global__ __launch_bounds__(256) void gemm_nt(
    const unsigned short* __restrict__ A,
    const unsigned short* __restrict__ W,
    const float* __restrict__ bias,
    unsigned short* __restrict__ outb,
    float* __restrict__ outf,
    int mode)
{
  __shared__ __align__(16) unsigned short wtile[2][64 * 32];

  const int lane  = threadIdx.x & 31;
  const int wave  = threadIdx.x >> 5;
  const int m0    = blockIdx.x * 256 + wave * 32;
  const int n0    = blockIdx.y * 64;
  const int r16   = lane & 15;
  const int khalf = lane >> 4;

  // LDS byte offsets for the TDM descriptors (this ptrtoint also makes the
  // buffer address escape, so the DS reads below can't be optimized away).
  const unsigned ldsoff0 = (unsigned)(uintptr_t)(lds_ushort*)&wtile[0][0];
  const unsigned ldsoff1 = (unsigned)(uintptr_t)(lds_ushort*)&wtile[1][0];

  v8f acc[2][4] = {};

  const unsigned short* arow0 = A + (size_t)(m0 + r16) * C_;
  const unsigned short* arow1 = arow0 + (size_t)16 * C_;
  const unsigned short* wbase = W + (size_t)n0 * C_;

  FragBF a0, a1, b0, b1;

  auto loadA = [&](FragBF& F0, FragBF& F1, int kk) {
    F0.q[0] = *(const uint4*)(arow0 + kk + khalf * 8);
    F0.q[1] = *(const uint4*)(arow0 + kk + 16 + khalf * 8);
    F1.q[0] = *(const uint4*)(arow1 + kk + khalf * 8);
    F1.q[1] = *(const uint4*)(arow1 + kk + 16 + khalf * 8);
  };

  // 8 WMMAs against one LDS-resident 64x32 W tile; B frags loaded up front.
  auto compute = [&](const FragBF& A0, const FragBF& A1,
                     const unsigned short* wt) {
    FragBF bf[4];
#pragma unroll
    for (int t = 0; t < 4; ++t) {
      const unsigned short* bp = wt + (t * 16 + r16) * 32 + khalf * 8;
      bf[t].q[0] = *(const uint4*)(bp);
      bf[t].q[1] = *(const uint4*)(bp + 16);
    }
#pragma unroll
    for (int t = 0; t < 4; ++t) {
      acc[0][t] = __builtin_amdgcn_wmma_f32_16x16x32_bf16(
          false, A0.v, false, bf[t].v, (short)0, acc[0][t], false, false);
      acc[1][t] = __builtin_amdgcn_wmma_f32_16x16x32_bf16(
          false, A1.v, false, bf[t].v, (short)0, acc[1][t], false, false);
    }
  };

  // Prologue: stage k=0 W tile + first A fragments.
  if (wave == 0) {
    tdm_load_w_tile(wbase, ldsoff0);
    __builtin_amdgcn_s_wait_tensorcnt(0);
  }
  loadA(a0, a1, 0);
  __syncthreads();

  for (int k = 0; k < C_; k += 64) {
    // Phase 0: compute from buf0/(a0,a1); DMA k+32 into buf1, prefetch A.
    if (wave == 0) tdm_load_w_tile(wbase + k + 32, ldsoff1);
    loadA(b0, b1, k + 32);
    compute(a0, a1, &wtile[0][0]);
    if (wave == 0) __builtin_amdgcn_s_wait_tensorcnt(0);
    __syncthreads();

    // Phase 1: compute from buf1/(b0,b1); DMA k+64 into buf0, prefetch A.
    if (k + 64 < C_) {
      if (wave == 0) tdm_load_w_tile(wbase + k + 64, ldsoff0);
      loadA(a0, a1, k + 64);
    }
    compute(b0, b1, &wtile[1][0]);
    if (wave == 0) __builtin_amdgcn_s_wait_tensorcnt(0);
    __syncthreads();
  }

  // Epilogue
#pragma unroll
  for (int g = 0; g < 2; ++g) {
    const int mrow_base = m0 + g * 16 + khalf * 8;
#pragma unroll
    for (int t = 0; t < 4; ++t) {
      const int col = n0 + t * 16 + r16;
      const float bv = bias[col];
#pragma unroll
      for (int i = 0; i < 8; ++i) {
        const int m = mrow_base + i;
        const float val = acc[g][t][i] + bv;
        if (mode == 2) {
          outf[(size_t)m * C_ + col] = val;
        } else {
          const int b = m >> 11;
          const int tt = m & (T_ - 1);
          const int h = col >> 6;
          const int d = col & 63;
          size_t idx;
          if (mode == 0)
            idx = (((size_t)(b * H_ + h)) * T_ + tt) * HD_ + d;
          else
            idx = (((size_t)(b * H_ + h)) * HD_ + d) * T_ + tt;
          outb[idx] = f32_to_bf16(val);
        }
      }
    }
  }
}

// ---------------------------------------------------------------------------
// Kernel 2: flash attention (causal, online softmax), prefetched.
//   Q,K: bf16 [B,H,T,HD]   Vt: bf16 [B,H,HD,T]   Ob: bf16 [B,T,C]
// One wave = 16 query rows x HD=64. kv chunks of 32; K frags ping-ponged,
// V frags issued before the S WMMAs so softmax overlaps the loads.
// ---------------------------------------------------------------------------
__global__ __launch_bounds__(256) void flash_attn(
    const unsigned short* __restrict__ Q,
    const unsigned short* __restrict__ K,
    const unsigned short* __restrict__ Vt,
    unsigned short* __restrict__ Ob)
{
  __shared__ __align__(16) unsigned short plds[8][16 * 32];

  const int lane  = threadIdx.x & 31;
  const int wave  = threadIdx.x >> 5;
  const int gw    = blockIdx.x * 8 + wave;
  const int qt    = gw & 127;
  const int bh    = gw >> 7;
  const int q0    = qt * 16;
  const int r16   = lane & 15;
  const int khalf = lane >> 4;

  const unsigned short* Qp = Q  + ((size_t)bh * T_ + q0) * HD_;
  const unsigned short* Kp = K  + (size_t)bh * T_ * HD_;
  const unsigned short* Vp = Vt + (size_t)bh * HD_ * T_;

  FragBF qa0, qa1;
  qa0.q[0] = *(const uint4*)(Qp + r16 * HD_ + khalf * 8);
  qa0.q[1] = *(const uint4*)(Qp + r16 * HD_ + 16 + khalf * 8);
  qa1.q[0] = *(const uint4*)(Qp + r16 * HD_ + 32 + khalf * 8);
  qa1.q[1] = *(const uint4*)(Qp + r16 * HD_ + 48 + khalf * 8);

  v8f o[4] = {};
  float mrow[8], lrow[8];
#pragma unroll
  for (int i = 0; i < 8; ++i) { mrow[i] = -__builtin_inff(); lrow[i] = 0.0f; }

  const int rowq = q0 + khalf * 8;
  const int njit = (q0 + 16 + 31) >> 5;
  unsigned short* pw = plds[wave];

  auto loadK = [&](int j, FragBF* kf) {
    const unsigned short* kr0 = Kp + (size_t)(j * 32 + r16) * HD_ + khalf * 8;
    const unsigned short* kr1 = kr0 + (size_t)16 * HD_;
    kf[0].q[0] = *(const uint4*)(kr0);      kf[0].q[1] = *(const uint4*)(kr0 + 16);
    kf[1].q[0] = *(const uint4*)(kr0 + 32); kf[1].q[1] = *(const uint4*)(kr0 + 48);
    kf[2].q[0] = *(const uint4*)(kr1);      kf[2].q[1] = *(const uint4*)(kr1 + 16);
    kf[3].q[0] = *(const uint4*)(kr1 + 32); kf[3].q[1] = *(const uint4*)(kr1 + 48);
  };

  auto step = [&](int j, FragBF* kc, FragBF* kn) {
    const int c0 = j * 32;
    const int c1 = c0 + 16;

    // V fragments issued first: in flight across the whole softmax phase.
    FragBF vb[4];
#pragma unroll
    for (int t = 0; t < 4; ++t) {
      const unsigned short* vr = Vp + (size_t)(t * 16 + r16) * T_ + c0 + khalf * 8;
      vb[t].q[0] = *(const uint4*)(vr);
      vb[t].q[1] = *(const uint4*)(vr + 16);
    }

    v8f s0 = {}, s1 = {};
    s0 = __builtin_amdgcn_wmma_f32_16x16x32_bf16(false, qa0.v, false, kc[0].v, (short)0, s0, false, false);
    s0 = __builtin_amdgcn_wmma_f32_16x16x32_bf16(false, qa1.v, false, kc[1].v, (short)0, s0, false, false);
    s1 = __builtin_amdgcn_wmma_f32_16x16x32_bf16(false, qa0.v, false, kc[2].v, (short)0, s1, false, false);
    s1 = __builtin_amdgcn_wmma_f32_16x16x32_bf16(false, qa1.v, false, kc[3].v, (short)0, s1, false, false);

    // Prefetch next K chunk while doing softmax.
    if (j + 1 < njit) loadK(j + 1, kn);

#pragma unroll
    for (int i = 0; i < 8; ++i) {
      const int qr = rowq + i;
      s0[i] = (c0 + r16 <= qr) ? s0[i] * 0.125f : -__builtin_inff();
      s1[i] = (c1 + r16 <= qr) ? s1[i] * 0.125f : -__builtin_inff();
    }

    float nm[8], corr[8];
#pragma unroll
    for (int i = 0; i < 8; ++i) {
      float v = fmaxf(s0[i], s1[i]);
      v = fmaxf(v, __shfl_xor(v, 1, 32));
      v = fmaxf(v, __shfl_xor(v, 2, 32));
      v = fmaxf(v, __shfl_xor(v, 4, 32));
      v = fmaxf(v, __shfl_xor(v, 8, 32));
      nm[i]   = fmaxf(mrow[i], v);
      corr[i] = __expf(mrow[i] - nm[i]);
      mrow[i] = nm[i];
    }

#pragma unroll
    for (int i = 0; i < 8; ++i) {
      const float p0 = __expf(s0[i] - nm[i]);
      const float p1 = __expf(s1[i] - nm[i]);
      lrow[i] = lrow[i] * corr[i] + p0 + p1;
      const int rr = (khalf * 8 + i) * 32;
      pw[rr + r16]      = f32_to_bf16(p0);
      pw[rr + 16 + r16] = f32_to_bf16(p1);
#pragma unroll
      for (int t = 0; t < 4; ++t) o[t][i] *= corr[i];
    }

    FragBF pa;
    pa.q[0] = *(const uint4*)(pw + r16 * 32 + khalf * 8);
    pa.q[1] = *(const uint4*)(pw + r16 * 32 + 16 + khalf * 8);

#pragma unroll
    for (int t = 0; t < 4; ++t)
      o[t] = __builtin_amdgcn_wmma_f32_16x16x32_bf16(
          false, pa.v, false, vb[t].v, (short)0, o[t], false, false);
  };

  FragBF kbuf0[4], kbuf1[4];
  loadK(0, kbuf0);
  int j = 0;
  while (j < njit) {
    step(j, kbuf0, kbuf1); ++j;
    if (j < njit) { step(j, kbuf1, kbuf0); ++j; }
  }

#pragma unroll
  for (int i = 0; i < 8; ++i) {
    float l = lrow[i];
    l += __shfl_xor(l, 1, 32);
    l += __shfl_xor(l, 2, 32);
    l += __shfl_xor(l, 4, 32);
    l += __shfl_xor(l, 8, 32);
    lrow[i] = 1.0f / l;
  }
  const int b = bh >> 4, h = bh & 15;
#pragma unroll
  for (int t = 0; t < 4; ++t) {
    const int col = h * 64 + t * 16 + r16;
#pragma unroll
    for (int i = 0; i < 8; ++i) {
      const int tt = q0 + khalf * 8 + i;
      Ob[((size_t)(b * T_ + tt)) * C_ + col] = f32_to_bf16(o[t][i] * lrow[i]);
    }
  }
}

// ---------------------------------------------------------------------------
// Launch: convert -> QKV GEMMs -> flash attention -> output projection
// ---------------------------------------------------------------------------
extern "C" void kernel_launch(void* const* d_in, const int* in_sizes, int n_in,
                              void* d_out, int out_size, void* d_ws, size_t ws_size,
                              hipStream_t stream) {
  const float* x  = (const float*)d_in[0];
  const float* Wq = (const float*)d_in[1];
  const float* bq = (const float*)d_in[2];
  const float* Wk = (const float*)d_in[3];
  const float* bk = (const float*)d_in[4];
  const float* Wv = (const float*)d_in[5];
  const float* bv = (const float*)d_in[6];
  const float* Wo = (const float*)d_in[7];
  const float* bo = (const float*)d_in[8];

  char* ws = (char*)d_ws;
  const size_t MB = 1024ull * 1024ull;
  unsigned short* xb    = (unsigned short*)(ws + 0 * MB);   // 16 MB [8192,1024]
  unsigned short* wqb   = (unsigned short*)(ws + 16 * MB);
  unsigned short* wkb   = (unsigned short*)(ws + 18 * MB);
  unsigned short* wvb   = (unsigned short*)(ws + 20 * MB);
  unsigned short* wob   = (unsigned short*)(ws + 22 * MB);
  unsigned short* Qb    = (unsigned short*)(ws + 24 * MB);  // [B,H,T,HD]
  unsigned short* Kb    = (unsigned short*)(ws + 40 * MB);  // [B,H,T,HD]
  unsigned short* Vt    = (unsigned short*)(ws + 56 * MB);  // [B,H,HD,T]
  unsigned short* attnb = (unsigned short*)(ws + 72 * MB);  // [B,T,C]

  const int nX = B_ * T_ * C_;
  const int nW = C_ * C_;

  cvt_f32_bf16<<<4096, 256, 0, stream>>>(x,  xb,  nX);
  cvt_f32_bf16<<<1024, 256, 0, stream>>>(Wq, wqb, nW);
  cvt_f32_bf16<<<1024, 256, 0, stream>>>(Wk, wkb, nW);
  cvt_f32_bf16<<<1024, 256, 0, stream>>>(Wv, wvb, nW);
  cvt_f32_bf16<<<1024, 256, 0, stream>>>(Wo, wob, nW);

  dim3 gg(32, 16);
  gemm_nt<<<gg, 256, 0, stream>>>(xb, wqb, bq, Qb, nullptr, 0);
  gemm_nt<<<gg, 256, 0, stream>>>(xb, wkb, bk, Kb, nullptr, 0);
  gemm_nt<<<gg, 256, 0, stream>>>(xb, wvb, bv, Vt, nullptr, 1);

  flash_attn<<<1024, 256, 0, stream>>>(Qb, Kb, Vt, attnb);

  gemm_nt<<<gg, 256, 0, stream>>>(attnb, wob, bo, nullptr, (float*)d_out, 2);
}
